// SAMGuidedCrossAttention_34797825032200
// MI455X (gfx1250) — matrix operand order repaired and checked by reading, not compiled, not run on hardware
//
#include <hip/hip_runtime.h>

typedef __attribute__((ext_vector_type(16))) _Float16 v16h;
typedef __attribute__((ext_vector_type(8)))  _Float16 v8h;
typedef __attribute__((ext_vector_type(8)))  float    v8f;
typedef __attribute__((ext_vector_type(4)))  unsigned int u32x4;
typedef __attribute__((ext_vector_type(8)))  int          i32x8;
typedef __attribute__((ext_vector_type(4)))  int          i32x4;

#if defined(__has_builtin)
#if __has_builtin(__builtin_amdgcn_tensor_load_to_lds)
#define HAVE_TDM 1
#endif
#endif

#define WMMA_F16(a, b, c) \
  __builtin_amdgcn_wmma_f32_16x16x32_f16(false, (a), false, (b), (short)0, (c), false, false)

// ---------------------------------------------------------------------------
// Fragment loaders (CDNA5 ISA 7.12.2, wave32). Layouts are chosen so each
// lane's 16 halfs are two contiguous 16-byte runs -> b128 loads.
//  A (16x32 f16): lane m = lane&15; half h -> K = (h&7) + (h>>3)*16 + (lane>>4)*8
//  B (32x16 f16): lane n = lane&15; half h -> K = (h&7) + (h>>3)*16 + (lane>>4)*8
//  C/D (16x16 f32): reg r -> M = r + (lane>>4)*8, N = lane&15
// ---------------------------------------------------------------------------

// A[m][k] row-major: element (m,k) at base[(row0+m)*ld + col0+k]
__device__ __forceinline__ v16h load_a_rm(const _Float16* base, int row0, int col0, int ld) {
  const int lane = threadIdx.x & 31;
  const int m  = lane & 15;
  const int kb = (lane >> 4) * 8;
  const _Float16* p = base + (size_t)(row0 + m) * ld + col0 + kb;
  v8h lo = *(const v8h*)(p);        // K = kb .. kb+7
  v8h hi = *(const v8h*)(p + 16);   // K = 16+kb .. 16+kb+7
  v16h a;
#pragma unroll
  for (int h = 0; h < 8; ++h) { a[h] = lo[h]; a[h + 8] = hi[h]; }
  return a;
}

// B[k][n] with element (k,n) at base[(col0+n)*ld + row0+k]  (contiguous in k)
__device__ __forceinline__ v16h load_b_cm(const _Float16* base, int row0, int col0, int ld) {
  const int lane = threadIdx.x & 31;
  const int n  = lane & 15;
  const int kb = (lane >> 4) * 8;
  const _Float16* p = base + (size_t)(col0 + n) * ld + row0 + kb;
  v8h lo = *(const v8h*)(p);
  v8h hi = *(const v8h*)(p + 16);
  v16h b;
#pragma unroll
  for (int h = 0; h < 8; ++h) { b[h] = lo[h]; b[h + 8] = hi[h]; }
  return b;
}

#if defined(HAVE_TDM)
// ---------------------------------------------------------------------------
// TDM: one DMA of a 3D tile (8 px contig, 8 rows stride 256, 256 ch stride 64K
// elements) from global f32 into contiguous LDS [c][p]. D# per ISA ch.8:
//   g0: count=1 | lds_addr | global_addr[56:0] | type=2
//   g1: data_size=2(4B); tensor_dim0/1 huge (no OOB); tile 8x8x256;
//       dim0_stride=256, dim1_stride=65536
//   g2: tensor_dim2 huge; g3/g4: zeros
// 6-arg builtin form (clang-23 / therock headers).
// ---------------------------------------------------------------------------
__device__ __forceinline__ void tdm_load_window(unsigned lds_addr, const float* gbase) {
  const unsigned long long ga = (unsigned long long)(size_t)gbase;
  u32x4 g0;
  g0[0] = 1u;                                            // count=1, user mode
  g0[1] = lds_addr;                                      // LDS byte address
  g0[2] = (unsigned)(ga & 0xFFFFFFFFu);                  // global_addr[31:0]
  g0[3] = (unsigned)((ga >> 32) & 0x1FFFFFFu) | (2u << 30);  // addr[56:32]|type=2
  i32x8 g1;
  g1[0] = 2 << 16;                                       // data_size = 4 bytes
  g1[1] = 0;                                             // tensor_dim0[15:0]=0 (lo of 2^30)
  g1[2] = 0x4000;                                        // tensor_dim0 = 2^30
  g1[3] = (8 << 16) | 0x4000;                            // tile_dim0=8 | tensor_dim1=2^30
  g1[4] = (256 << 16) | 8;                               // tile_dim2=256 | tile_dim1=8
  g1[5] = 256;                                           // tensor_dim0_stride = W
  g1[6] = 0;                                             // strides mid bits
  g1[7] = 1;                                             // tensor_dim1_stride = 65536
  i32x4 g2;
  g2[0] = 0x40000000;                                    // tensor_dim2 = 2^30
  g2[1] = 0; g2[2] = 0; g2[3] = 0;
  i32x4 g3;
  g3[0] = 0; g3[1] = 0; g3[2] = 0; g3[3] = 0;
  i32x8 g4;
#pragma unroll
  for (int i = 0; i < 8; ++i) g4[i] = 0;
  __builtin_amdgcn_tensor_load_to_lds(g0, g1, g2, g3, g4, 0);
}
#else
// Fallback: per-lane async copy of 16 contiguous bytes global -> LDS.
__device__ __forceinline__ void async_g2l_b128(unsigned lds_byte_off, const float* gptr) {
  asm volatile("global_load_async_to_lds_b128 %0, %1, off"
               :: "v"(lds_byte_off), "v"(gptr) : "memory");
}
#endif

// ---------------------------------------------------------------------------
// Projection helper: Y[256co x 64px] = W[256x256] * X[px-major 64x256] + bias.
// Each wave owns 2 co-tiles; A-fragments hoisted and reused over 4 px-tiles.
// PR: 0 = Q (pixel-major dst, *scale), 1 = K (pixel-major), 2 = V (chan-major)
// ---------------------------------------------------------------------------
template <int PR>
__device__ __forceinline__ void proj_qkv(const _Float16* __restrict__ W,
                                         const _Float16* __restrict__ X,
                                         const float* __restrict__ bias,
                                         _Float16* __restrict__ dst,
                                         int wave, int lane, float mul) {
  const int mb = (lane >> 4) * 8, n = lane & 15;
  for (int ct = 0; ct < 2; ++ct) {
    const int co0 = (wave * 2 + ct) * 16;
    v16h a[8];
#pragma unroll
    for (int kk = 0; kk < 8; ++kk) a[kk] = load_a_rm(W, co0, kk * 32, 256);
    v8f binit;
#pragma unroll
    for (int r = 0; r < 8; ++r) binit[r] = bias[co0 + mb + r];
    for (int pt = 0; pt < 4; ++pt) {
      v8f acc = binit;                                  // bias folded into C
#pragma unroll
      for (int kk = 0; kk < 8; ++kk) {
        v16h bm = load_b_cm(X, kk * 32, pt * 16, 256);  // X[p][cin], ds b128
        acc = WMMA_F16(a[kk], bm, acc);
      }
#pragma unroll
      for (int r = 0; r < 8; ++r) {
        const int co = co0 + mb + r;
        float v = acc[r];
        if (PR == 0) v *= mul;                          // attn scale folded into Q
        if (PR < 2) dst[(pt * 16 + n) * 256 + co] = (_Float16)v;
        else        dst[co * 64 + pt * 16 + n]    = (_Float16)v;
      }
    }
  }
}

// ---------------------------------------------------------------------------
// Kernel 0: convert the four 256x256 f32 weight matrices to f16 in workspace.
// ---------------------------------------------------------------------------
__global__ void cvt_weights_kernel(const float* __restrict__ wq, const float* __restrict__ wk,
                                   const float* __restrict__ wv, const float* __restrict__ wp,
                                   _Float16* __restrict__ dst) {
  int i = blockIdx.x * blockDim.x + threadIdx.x;
  if (i >= 4 * 65536) return;
  int m = i >> 16, j = i & 65535;
  const float* src = (m == 0) ? wq : (m == 1) ? wk : (m == 2) ? wv : wp;
  dst[i] = (_Float16)src[j];
}

// ---------------------------------------------------------------------------
// Kernel 1: fully fused per-window attention. block = one 8x8 window, 8 waves.
// LDS map (bytes, 192 KB total):
//   phase 0 : sXq32[0,64K) sXk32[64K,128K)  sXq16[128K,160K) sXk16[160K,192K)
//   phase 1+: sQ[0,32K) sK[32K,64K) sV[64K,96K) sO[96K,128K)   (alias X32, dead)
//   phase 2+: sP[128K,192K)                                    (alias X16, dead)
// f16 layouts: X/Q/K/O pixel-major [64][256], V channel-major [256][64],
//              P row-major [64][64] per head.
// ---------------------------------------------------------------------------
__global__ void __launch_bounds__(256, 1)
win_attn_kernel(const float* __restrict__ qf, const float* __restrict__ kf,
                const _Float16* __restrict__ w16,
                const float* __restrict__ bq, const float* __restrict__ bk,
                const float* __restrict__ bv, const float* __restrict__ bp,
                float* __restrict__ out) {
  extern __shared__ char smb[];
  float*    sXq32 = (float*)(smb);
  float*    sXk32 = (float*)(smb + 65536);
  _Float16* sXq16 = (_Float16*)(smb + 131072);
  _Float16* sXk16 = (_Float16*)(smb + 163840);
  _Float16* sQ    = (_Float16*)(smb);
  _Float16* sK    = (_Float16*)(smb + 32768);
  _Float16* sV    = (_Float16*)(smb + 65536);
  _Float16* sO    = (_Float16*)(smb + 98304);
  _Float16* sP    = (_Float16*)(smb + 131072);

  const int tid  = threadIdx.x;
  const int wave = tid >> 5;
  const int lane = tid & 31;

  const int win = blockIdx.x;
  const int b   = win >> 10;
  const int wy  = (win >> 5) & 31;
  const int wx  = win & 31;
  const int h0  = wy * 8, w0 = wx * 8;

  // ---- phase 0a: DMA both window patches (f32) into LDS ----
#if defined(HAVE_TDM)
  if (wave == 0) {
    const size_t base = (size_t)b * 256 * 65536 + (size_t)h0 * 256 + w0;
    tdm_load_window(0,     qf + base);          // TENSORcnt-tracked tile DMA
    tdm_load_window(65536, kf + base);
    __builtin_amdgcn_s_wait_tensorcnt(0);
  }
#else
  for (int i = tid; i < 4096; i += 256) {
    int c = i >> 4;
    int p = (i & 15) * 4;                 // 4 contiguous pixels along W = 16 bytes
    int py = p >> 3, px = p & 7;
    size_t gi = ((size_t)(b * 256 + c) * 256 + (h0 + py)) * 256 + (w0 + px);
    unsigned loff = (unsigned)((c * 64 + p) * 4);
    async_g2l_b128(loff,         qf + gi);
    async_g2l_b128(loff + 65536, kf + gi);
  }
  asm volatile("s_wait_asynccnt 0x0" ::: "memory");
#endif
  __syncthreads();

  // ---- phase 0b: f32 -> f16 with [c][p] -> [p][c] transpose (LDS->LDS) ----
  for (int i = tid; i < 4096; i += 256) {
    int c = i >> 4;
    int p = (i & 15) * 4;
    float4 aq = *(const float4*)(sXq32 + c * 64 + p);
    float4 ak = *(const float4*)(sXk32 + c * 64 + p);
    sXq16[(p + 0) * 256 + c] = (_Float16)aq.x; sXq16[(p + 1) * 256 + c] = (_Float16)aq.y;
    sXq16[(p + 2) * 256 + c] = (_Float16)aq.z; sXq16[(p + 3) * 256 + c] = (_Float16)aq.w;
    sXk16[(p + 0) * 256 + c] = (_Float16)ak.x; sXk16[(p + 1) * 256 + c] = (_Float16)ak.y;
    sXk16[(p + 2) * 256 + c] = (_Float16)ak.z; sXk16[(p + 3) * 256 + c] = (_Float16)ak.w;
  }
  __syncthreads();

  const float scale = 0.17677669529663687f;    // (C/heads)^-0.5 = 32^-0.5

  // ---- phase 1: Q/K/V projections (fully specialized, A-fragments hoisted) ----
  proj_qkv<0>(w16,               sXq16, bq, sQ, wave, lane, scale);
  proj_qkv<1>(w16 + 1 * 65536,   sXk16, bk, sK, wave, lane, 1.0f);
  proj_qkv<2>(w16 + 2 * 65536,   sXk16, bv, sV, wave, lane, 1.0f);
  __syncthreads();

  // ---- phase 2: attention, wave == head ----
  {
    const int hd = wave;
    _Float16* Ph = sP + hd * 4096;                     // P[q][k] row-major

    for (int qi = 0; qi < 4; ++qi) {
      const int q0 = qi * 16;
      v16h a = load_a_rm(sQ, q0, hd * 32, 256);        // A[q][d] = Q[q][hd*32+d]
      v8f sc[4];
#pragma unroll
      for (int ki = 0; ki < 4; ++ki) {
        v16h bm = load_b_cm(sK, hd * 32, ki * 16, 256); // B[d][k] = K[kpix][hd*32+d]
        v8f z = {};
        sc[ki] = WMMA_F16(a, bm, z);                   // K-dim = 32: one WMMA/tile
      }
      // softmax over 64 cols per row; rows live in 16-lane halves -> width-16 shfl
#pragma unroll
      for (int r = 0; r < 8; ++r) {
        float m = fmaxf(fmaxf(sc[0][r], sc[1][r]), fmaxf(sc[2][r], sc[3][r]));
        m = fmaxf(m, __shfl_xor(m, 1, 16));
        m = fmaxf(m, __shfl_xor(m, 2, 16));
        m = fmaxf(m, __shfl_xor(m, 4, 16));
        m = fmaxf(m, __shfl_xor(m, 8, 16));
        float s = 0.f;
#pragma unroll
        for (int ki = 0; ki < 4; ++ki) {
          float e = __expf(sc[ki][r] - m);
          sc[ki][r] = e;
          s += e;
        }
        s += __shfl_xor(s, 1, 16);
        s += __shfl_xor(s, 2, 16);
        s += __shfl_xor(s, 4, 16);
        s += __shfl_xor(s, 8, 16);
        const float inv = 1.0f / s;
#pragma unroll
        for (int ki = 0; ki < 4; ++ki) sc[ki][r] *= inv;
      }
      const int mb = (lane >> 4) * 8, n = lane & 15;
#pragma unroll
      for (int ki = 0; ki < 4; ++ki)
#pragma unroll
        for (int r = 0; r < 8; ++r)
          Ph[(q0 + r + mb) * 64 + ki * 16 + n] = (_Float16)sc[ki][r];
    }
    // O[q][d] = sum_k P[q][k] * V[d][k]  (same-wave LDS RAW, DScnt-ordered)
    for (int qi = 0; qi < 4; ++qi) {
      for (int di = 0; di < 2; ++di) {
        v8f acc = {};
#pragma unroll
        for (int kk = 0; kk < 2; ++kk) {
          v16h a  = load_a_rm(Ph, qi * 16, kk * 32, 64);             // P row-major
          v16h bm = load_b_cm(sV, kk * 32, hd * 32 + di * 16, 64);   // V[c][kpix]
          acc = WMMA_F16(a, bm, acc);
        }
        const int mb = (lane >> 4) * 8, n = lane & 15;
#pragma unroll
        for (int r = 0; r < 8; ++r)      // O pixel-major for the out projection
          sO[(qi * 16 + r + mb) * 256 + hd * 32 + di * 16 + n] = (_Float16)acc[r];
      }
    }
  }
  __syncthreads();

  // ---- phase 3: output projection Wp * O + bp, f32 to global ----
  {
    const _Float16* W = w16 + (size_t)3 * 65536;
    const int mb = (lane >> 4) * 8, n = lane & 15;
    for (int ct = 0; ct < 2; ++ct) {
      const int co0 = (wave * 2 + ct) * 16;
      v16h a[8];
#pragma unroll
      for (int kk = 0; kk < 8; ++kk) a[kk] = load_a_rm(W, co0, kk * 32, 256);
      v8f binit;
#pragma unroll
      for (int r = 0; r < 8; ++r) binit[r] = bp[co0 + mb + r];
      for (int pt = 0; pt < 4; ++pt) {
        v8f acc = binit;
#pragma unroll
        for (int kk = 0; kk < 8; ++kk) {
          v16h bm = load_b_cm(sO, kk * 32, pt * 16, 256);  // O[p][c], ds b128
          acc = WMMA_F16(a[kk], bm, acc);
        }
        const int p = pt * 16 + n, py = p >> 3, px = p & 7;
#pragma unroll
        for (int r = 0; r < 8; ++r) {
          const int co = co0 + mb + r;
          size_t gi = ((size_t)(b * 256 + co) * 256 + (h0 + py)) * 256 + (w0 + px);
          out[gi] = acc[r];
        }
      }
    }
  }
}

// ---------------------------------------------------------------------------
extern "C" void kernel_launch(void* const* d_in, const int* in_sizes, int n_in,
                              void* d_out, int out_size, void* d_ws, size_t ws_size,
                              hipStream_t stream) {
  (void)in_sizes; (void)n_in; (void)out_size; (void)ws_size;
  const float* qf = (const float*)d_in[0];
  const float* kf = (const float*)d_in[1];
  const float* wq = (const float*)d_in[2];
  const float* bq = (const float*)d_in[3];
  const float* wk = (const float*)d_in[4];
  const float* bk = (const float*)d_in[5];
  const float* wv = (const float*)d_in[6];
  const float* bv = (const float*)d_in[7];
  const float* wp = (const float*)d_in[8];
  const float* bp = (const float*)d_in[9];
  float* out = (float*)d_out;

  _Float16* w16 = (_Float16*)d_ws;     // 4 * 65536 halfs = 512 KB (L2-resident)

  cvt_weights_kernel<<<(4 * 65536 + 255) / 256, 256, 0, stream>>>(wq, wk, wv, wp, w16);

  const int lds_bytes = 196608;        // 192 KB of the 320 KB WGP LDS
  (void)hipFuncSetAttribute((const void*)win_attn_kernel,
                            hipFuncAttributeMaxDynamicSharedMemorySize, lds_bytes);
  win_attn_kernel<<<4096, 256, lds_bytes, stream>>>(qf, kf, w16, bq, bk, bv, bp, out);
}
